// VectorQuantizer_55310588838386
// MI455X (gfx1250) — matrix operand order repaired and checked by reading, not compiled, not run on hardware
//
#include <hip/hip_runtime.h>
#include <hip/hip_bf16.h>

typedef __attribute__((ext_vector_type(16))) _Float16 v16h;
typedef __attribute__((ext_vector_type(8)))  _Float16 v8h;
typedef __attribute__((ext_vector_type(8)))  float    v8f;

#define DIM     128
#define HWSZ    1024
#define KCODES  1024
#define NVEC    65536
#define TILE_N  128
#define WCHUNK  32          // codes per staged chunk (double-buffered)
#define NCHUNK  (KCODES / WCHUNK)
#define LDSPAD  8

// ---- LDS byte-address helper (ptrtoint of an addrspace(3) pointer) ----
static __device__ __forceinline__ unsigned lds_addr(void* p) {
  return (unsigned)(size_t)(__attribute__((address_space(3))) void*)p;
}

// ---- CDNA5 async global->LDS 16B copy (ASYNCcnt-tracked) ----
static __device__ __forceinline__ void async_copy_b128(const void* g, void* l) {
  unsigned laddr = lds_addr(l);
  asm volatile("global_load_async_to_lds_b128 %0, %1, off"
               :: "v"(laddr), "v"(g) : "memory");
}
static __device__ __forceinline__ void wait_async0() {
  asm volatile("s_wait_asynccnt 0" ::: "memory");
}

// ---------------------------------------------------------------- kernel 0
// Per code k: wnorm[k] = ||W_k||^2, and Wh[k][:] = f16(W_k) for async staging.
__global__ __launch_bounds__(256) void vq_prep_kernel(
    const float* __restrict__ Wm, _Float16* __restrict__ Wh,
    float* __restrict__ wnorm) {
  int k = blockIdx.x * 256 + threadIdx.x;
  if (k >= KCODES) return;
  const float* row = Wm + (size_t)k * DIM;
  _Float16* hrow = Wh + (size_t)k * DIM;
  float s = 0.f;
#pragma unroll 8
  for (int d = 0; d < DIM; ++d) {
    float v = row[d];
    s += v * v;
    hrow[d] = (_Float16)v;
  }
  wnorm[k] = s;
}

// ---------------------------------------------------------------- kernel 1
// 512 blocks x 256 threads (8 waves). Block: 128 rows x all 1024 codes.
// W chunks staged with global_load_async_to_lds_b128 into a double buffer.
__global__ __launch_bounds__(256) void vq_argmin_kernel(
    const float* __restrict__ z, const _Float16* __restrict__ Wh,
    const float* __restrict__ wnorm, float* __restrict__ idx_out) {
  __shared__ _Float16 Zl[TILE_N][DIM + LDSPAD];
  __shared__ _Float16 Wl[2][WCHUNK][DIM + LDSPAD];

  const int t   = threadIdx.x;
  const int n0  = blockIdx.x * TILE_N;
  const int b   = n0 >> 10;                 // tile stays within one batch
  const int hw0 = n0 & (HWSZ - 1);
  const float* zb = z + ((size_t)b * DIM) * HWSZ + hw0;

  // ---- stage z tile (coalesced over n) into f16 LDS ----
  const int nrow  = t & (TILE_N - 1);
  const int dhalf = t >> 7;
#pragma unroll
  for (int it = 0; it < 64; ++it) {
    int d = it * 2 + dhalf;
    Zl[nrow][d] = (_Float16)zb[(size_t)d * HWSZ + nrow];
  }

  // ---- kick off async stage of chunk 0 while z barrier settles ----
  // chunk stage: WCHUNK rows x 256B = 8KB -> 512 x 16B ops -> 2 per thread
#pragma unroll
  for (int it = 0; it < 2; ++it) {
    int i = it * 256 + t;
    int row = i >> 4, seg = i & 15;
    async_copy_b128(Wh + ((size_t)row * DIM + seg * 8), &Wl[0][row][seg * 8]);
  }
  __syncthreads();

  const int lane = t & 31;
  const int wv   = t >> 5;
  const int lm   = lane & 15;
  const int lh   = lane >> 4;
  const int arow = wv * 16 + lm;

  // ---- preload A fragments (ISA 16-bit A 16x32 layout), 4 K-chunks ----
  v16h a[4];
#pragma unroll
  for (int dc = 0; dc < 4; ++dc) {
    const v8h lo = *(const v8h*)(&Zl[arow][dc * 32 + lh * 8]);       // K base+0..7
    const v8h hi = *(const v8h*)(&Zl[arow][dc * 32 + 16 + lh * 8]);  // K base+16..23
#pragma unroll
    for (int j = 0; j < 8; ++j) { a[dc][j] = lo[j]; a[dc][j + 8] = hi[j]; }
  }

  float best[8];
  int   bix[8];
#pragma unroll
  for (int v = 0; v < 8; ++v) { best[v] = 3.4e38f; bix[v] = 0; }

  for (int ch = 0; ch < NCHUNK; ++ch) {
    const int cur = ch & 1;
    wait_async0();                 // this wave's loads for buf[cur] complete
    __syncthreads();               // all waves' loads done; prev compute done

    if (ch + 1 < NCHUNK) {         // prefetch next chunk into other buffer
#pragma unroll
      for (int it = 0; it < 2; ++it) {
        int i = it * 256 + t;
        int row = i >> 4, seg = i & 15;
        async_copy_b128(Wh + ((size_t)((ch + 1) * WCHUNK + row) * DIM + seg * 8),
                        &Wl[cur ^ 1][row][seg * 8]);
      }
    }

#pragma unroll
    for (int cg = 0; cg < WCHUNK / 16; ++cg) {
      v8f c = {};
#pragma unroll
      for (int dc = 0; dc < 4; ++dc) {
        const int brow = cg * 16 + lm;          // B column = code (lane%16)
        const v8h lo = *(const v8h*)(&Wl[cur][brow][dc * 32 + lh * 8]);
        const v8h hi = *(const v8h*)(&Wl[cur][brow][dc * 32 + 16 + lh * 8]);
        v16h bf;
#pragma unroll
        for (int j = 0; j < 8; ++j) { bf[j] = lo[j]; bf[j + 8] = hi[j]; }
        c = __builtin_amdgcn_wmma_f32_16x16x32_f16(false, a[dc], false, bf,
                                                   (short)0, c, false, false);
      }
      const int   code = ch * WCHUNK + cg * 16 + lm;
      const float w2   = wnorm[code];           // L2-resident, 64B per cg
#pragma unroll
      for (int v = 0; v < 8; ++v) {
        float s = w2 - 2.0f * c[v];             // ||w||^2 - 2 z.w
        if (s < best[v]) { best[v] = s; bix[v] = code; }
      }
    }
  }

  // ---- min/argmin across the 16 lanes of each half (C layout: lane=N) ----
#pragma unroll
  for (int v = 0; v < 8; ++v) {
    float dmin = best[v];
    int   imin = bix[v];
#pragma unroll
    for (int off = 1; off < 16; off <<= 1) {
      float od = __shfl_xor(dmin, off, 32);
      int   oi = __shfl_xor(imin, off, 32);
      if (od < dmin || (od == dmin && oi < imin)) { dmin = od; imin = oi; }
    }
    if (lm == 0) {
      int row = wv * 16 + v + lh * 8;
      idx_out[n0 + row] = (float)imin;
    }
  }
}

// ---------------------------------------------------------------- kernel 2
__global__ __launch_bounds__(256) void vq_quant_kernel(
    const float* __restrict__ z, const float* __restrict__ Wm,
    const float* __restrict__ idxf, float* __restrict__ out,
    float* __restrict__ counts, float* __restrict__ sumacc) {
  __shared__ float red[256];
  const int t     = threadIdx.x;
  const int n0    = blockIdx.x * TILE_N;
  const int nrow  = t & (TILE_N - 1);
  const int dhalf = t >> 7;
  const int n     = n0 + nrow;
  const int code  = (int)idxf[n];
  const float* wrow = Wm + (size_t)code * DIM;
  const int b  = n >> 10;
  const int hw = n & (HWSZ - 1);
  const float* zbp = z   + ((size_t)b * DIM) * HWSZ + hw;
  float*       obp = out + ((size_t)b * DIM) * HWSZ + hw;

  float acc = 0.f;
#pragma unroll
  for (int it = 0; it < 64; ++it) {
    int d = it * 2 + dhalf;
    float q  = wrow[d];
    float zv = zbp[(size_t)d * HWSZ];
    obp[(size_t)d * HWSZ] = q;                  // z + sg(q - z) == q
    float df = q - zv;
    acc += df * df;
  }

  red[t] = acc;
  __syncthreads();
  for (int s = 128; s > 0; s >>= 1) {
    if (t < s) red[t] += red[t + s];
    __syncthreads();
  }
  if (t == 0) atomicAdd(sumacc, red[0]);
  if (t < TILE_N) atomicAdd(&counts[code], 1.0f);
}

// ---------------------------------------------------------------- kernel 3
__global__ __launch_bounds__(1024) void vq_scalars_kernel(
    const float* __restrict__ counts, const float* __restrict__ sumacc,
    float* __restrict__ out) {
  __shared__ float red[1024];
  const int t = threadIdx.x;
  float p = counts[t] * (1.0f / (float)NVEC);
  red[t] = p * __logf(p + 1e-10f);
  __syncthreads();
  for (int s = 512; s > 0; s >>= 1) {
    if (t < s) red[t] += red[t + s];
    __syncthreads();
  }
  if (t == 0) {
    float perp = __expf(-red[0]);
    float cb   = sumacc[0] * (1.0f / 8388608.0f);
    out[8388608] = 1.25f * cb;   // vq_loss_total
    out[8388609] = cb;           // codebook_loss
    out[8388610] = 0.25f * cb;   // commit_loss
    out[8388611] = perp;         // perplexity
  }
}

// ---------------------------------------------------------------- launch
extern "C" void kernel_launch(void* const* d_in, const int* in_sizes, int n_in,
                              void* d_out, int out_size, void* d_ws, size_t ws_size,
                              hipStream_t stream) {
  (void)in_sizes; (void)n_in; (void)out_size; (void)ws_size;
  const float* z  = (const float*)d_in[0];
  const float* Wm = (const float*)d_in[1];
  float* out = (float*)d_out;

  _Float16* Wh  = (_Float16*)d_ws;                              // 256 KB
  float* wnorm  = (float*)((char*)d_ws + (size_t)KCODES * DIM * 2);
  float* counts = wnorm + KCODES;
  float* sumacc = counts + KCODES;

  float* idx_out = out + 8388608 + 4;   // idx region (float-encoded)

  hipMemsetAsync(counts, 0, (KCODES + 1) * sizeof(float), stream);

  vq_prep_kernel  <<<KCODES / 256, 256, 0, stream>>>(Wm, Wh, wnorm);
  vq_argmin_kernel<<<NVEC / TILE_N, 256, 0, stream>>>(z, Wh, wnorm, idx_out);
  vq_quant_kernel <<<NVEC / TILE_N, 256, 0, stream>>>(z, Wm, idx_out, out, counts, sumacc);
  vq_scalars_kernel<<<1, 1024, 0, stream>>>(counts, sumacc, out);
}
